// MolHyGAN_31653908971673
// MI455X (gfx1250) — compile-verified, matrix operand order checked by reading, not run on hardware
//
#include <hip/hip_runtime.h>

// ---------------- problem constants (from reference) ----------------
#define N_HE     50000
#define N_EDGES  1200000
#define IN_FEAT  200
#define HE_DIM   512
#define E_DIM    128
#define Q_DIM    64

typedef __attribute__((ext_vector_type(16))) _Float16 v16h;
typedef __attribute__((ext_vector_type(2)))  _Float16 v2h;
typedef __attribute__((ext_vector_type(8)))  float    v8f;

// ---------------- tiled WMMA GEMM: C = act(A[M,K] @ W[K,N] + bias) ----------------
// A row-major (lda), W row-major KxN (ldw), C row-major (ldc).
#define BM 128
#define BN 128
#define BK 32
#define LDT 40   // padded LDS row stride (elements); 80B rows -> 16B aligned frag loads

__global__ __launch_bounds__(256)
void gemm_wmma_bias_act(const float* __restrict__ A, int lda,
                        const float* __restrict__ W, int ldw,
                        const float* __restrict__ bias,
                        float* __restrict__ C, int ldc,
                        int M, int N, int K, int act)
{
    __shared__ __align__(16) _Float16 sA[BM * LDT];
    __shared__ __align__(16) _Float16 sB[BN * LDT];   // holds W^T tile: [n][k]

    const int t    = threadIdx.x;
    const int lane = t & 31;
    const int wid  = t >> 5;
    const int wm   = wid & 3;    // wave row -> 32 rows
    const int wn   = wid >> 2;   // wave col -> 64 cols
    const int m0   = blockIdx.y * BM;
    const int n0   = blockIdx.x * BN;
    const int half = lane >> 4;
    const int l16  = lane & 15;

    v8f zero = {};
    v8f acc[2][4];
    for (int i = 0; i < 2; ++i)
        for (int j = 0; j < 4; ++j) acc[i][j] = zero;

    union Frag { v16h v; uint4 u[2]; };

    const int nk = (K + BK - 1) / BK;
    for (int kt = 0; kt < nk; ++kt) {
        const int k0 = kt * BK;

        // ---- batched, unpredicated staging: clamp addresses, mask in registers ----
        // A tile: 2048 k-pairs, 8 pairs (16 floats) per thread.
        float av[16];
        #pragma unroll
        for (int s = 0; s < 8; ++s) {
            int idx = t + s * 256;
            int r  = idx >> 4;
            int c  = (idx & 15) * 2;
            int gm = m0 + r, gk = k0 + c;
            int cm  = gm < M ? gm : M - 1;
            int ck0 = gk     < K ? gk     : K - 1;
            int ck1 = gk + 1 < K ? gk + 1 : K - 1;
            av[2 * s]     = A[(size_t)cm * lda + ck0];
            av[2 * s + 1] = A[(size_t)cm * lda + ck1];
        }
        // W^T tile: 2048 k-pairs, 8 pairs per thread (coalesced along n).
        float bv[16];
        #pragma unroll
        for (int s = 0; s < 8; ++s) {
            int idx = t + s * 256;
            int kp = idx >> 7;
            int n  = idx & 127;
            int gk = k0 + kp * 2, gn = n0 + n;
            int cn  = gn < N ? gn : N - 1;
            int ck0 = gk     < K ? gk     : K - 1;
            int ck1 = gk + 1 < K ? gk + 1 : K - 1;
            bv[2 * s]     = W[(size_t)ck0 * ldw + cn];
            bv[2 * s + 1] = W[(size_t)ck1 * ldw + cn];
        }
        // Prefetch next K-tile while converting/storing this one.
        if (kt + 1 < nk) {
            int pm = m0 + (t >> 1);          pm = pm < M ? pm : M - 1;
            int pk = k0 + BK + (t & 1) * 16; pk = pk < K ? pk : K - 1;
            __builtin_prefetch(&A[(size_t)pm * lda + pk], 0, 0);
            int qk = k0 + BK + (t >> 7) * 16;  qk = qk < K ? qk : K - 1;
            int qn = n0 + (t & 127);           qn = qn < N ? qn : N - 1;
            __builtin_prefetch(&W[(size_t)qk * ldw + qn], 0, 0);
        }
        // Mask + convert + packed b32 LDS stores.
        #pragma unroll
        for (int s = 0; s < 8; ++s) {
            int idx = t + s * 256;
            int r  = idx >> 4;
            int c  = (idx & 15) * 2;
            int gm = m0 + r, gk = k0 + c;
            float a0 = (gm < M && gk     < K) ? av[2 * s]     : 0.0f;
            float a1 = (gm < M && gk + 1 < K) ? av[2 * s + 1] : 0.0f;
            v2h pk; pk.x = (_Float16)a0; pk.y = (_Float16)a1;
            *(v2h*)&sA[r * LDT + c] = pk;
        }
        #pragma unroll
        for (int s = 0; s < 8; ++s) {
            int idx = t + s * 256;
            int kp = idx >> 7;
            int n  = idx & 127;
            int gk = k0 + kp * 2, gn = n0 + n;
            float b0 = (gn < N && gk     < K) ? bv[2 * s]     : 0.0f;
            float b1 = (gn < N && gk + 1 < K) ? bv[2 * s + 1] : 0.0f;
            v2h pk; pk.x = (_Float16)b0; pk.y = (_Float16)b1;
            *(v2h*)&sB[n * LDT + kp * 2] = pk;
        }
        __syncthreads();

        // ---- fragments per 16-bit A/B VGPR layout (lanes<16: K 0..7,16..23) ----
        Frag af[2], bf[4];
        for (int i = 0; i < 2; ++i) {
            const _Float16* p = &sA[(wm * 32 + i * 16 + l16) * LDT + half * 8];
            af[i].u[0] = *(const uint4*)p;
            af[i].u[1] = *(const uint4*)(p + 16);
        }
        for (int j = 0; j < 4; ++j) {
            const _Float16* p = &sB[(wn * 64 + j * 16 + l16) * LDT + half * 8];
            bf[j].u[0] = *(const uint4*)p;
            bf[j].u[1] = *(const uint4*)(p + 16);
        }
        for (int i = 0; i < 2; ++i)
            for (int j = 0; j < 4; ++j)
                acc[i][j] = __builtin_amdgcn_wmma_f32_16x16x32_f16(
                    false, af[i].v, false, bf[j].v, (short)0, acc[i][j], false, false);
        __syncthreads();
    }

    // ---- epilogue: bias + activation + store (C/D layout: vgpr r -> M=r + 8*half) ----
    for (int i = 0; i < 2; ++i) {
        int mbase = m0 + wm * 32 + i * 16 + half * 8;
        for (int j = 0; j < 4; ++j) {
            int n = n0 + wn * 64 + j * 16 + l16;
            if (n >= N) continue;
            float b = bias ? bias[n] : 0.0f;
            for (int r = 0; r < 8; ++r) {
                int m = mbase + r;
                if (m < M) {
                    float v = acc[i][j][r] + b;
                    if (act) v = fmaxf(v, 0.0f);
                    C[(size_t)m * ldc + n] = v;
                }
            }
        }
    }
}

// ---------------- fused bias: b12 = b1 @ w2 + b2 ----------------
__global__ void fuse_bias_kernel(const float* __restrict__ w1b,
                                 const float* __restrict__ w2w,
                                 const float* __restrict__ w2b,
                                 float* __restrict__ b12)
{
    int n = threadIdx.x;   // 128
    float s = w2b[n];
    for (int k = 0; k < HE_DIM; ++k) s += w1b[k] * w2w[k * E_DIM + n];
    b12[n] = s;
}

// ---------------- CSR build ----------------
__global__ void zero_i32(int* p, int n) {
    int i = blockIdx.x * 256 + threadIdx.x;
    if (i < n) p[i] = 0;
}

__global__ void count_edges(const int* __restrict__ seg, int n, int* __restrict__ cnt) {
    int i = blockIdx.x * 256 + threadIdx.x;
    if (i < n) atomicAdd(&cnt[seg[i]], 1);
}

__global__ __launch_bounds__(1024)
void scan_offsets(const int* __restrict__ cnt, int n,
                  int* __restrict__ offs, int* __restrict__ cursor)
{
    __shared__ int buf[1024];
    __shared__ int carry;
    int t = threadIdx.x;
    if (t == 0) carry = 0;
    __syncthreads();
    for (int base = 0; base < n; base += 1024) {
        int i = base + t;
        int v = (i < n) ? cnt[i] : 0;
        buf[t] = v;
        __syncthreads();
        for (int off = 1; off < 1024; off <<= 1) {
            int x = (t >= off) ? buf[t - off] : 0;
            __syncthreads();
            buf[t] += x;
            __syncthreads();
        }
        int excl = buf[t] - v + carry;
        if (i < n) { offs[i] = excl; cursor[i] = excl; }
        __syncthreads();
        if (t == 1023) carry += buf[1023];
        __syncthreads();
    }
    if (t == 0) offs[n] = carry;
}

__global__ void fill_csr(const int* __restrict__ seg, const int* __restrict__ other,
                         int n, int* __restrict__ cursor, int* __restrict__ csr)
{
    int i = blockIdx.x * 256 + threadIdx.x;
    if (i < n) {
        int pos = atomicAdd(&cursor[seg[i]], 1);
        csr[pos] = other[i];
    }
}

// ---------------- segment-softmax attention: one wave32 per hyperedge ----------------
// att[e] = leaky_relu(<k2[d], qfn[h]> / 8); softmax over edges of h; out = sum ex*v2[d] / sum ex
__global__ __launch_bounds__(256)
void seg_attn_wave(const float* __restrict__ qfn,   // [N_HE, 128]
                   const float* __restrict__ k2,    // [n_t, 128]
                   const float* __restrict__ v2,    // [n_t, 128]
                   const int*   __restrict__ offs,  // [N_HE+1]
                   const int*   __restrict__ csr,   // node id per edge, grouped by h
                   float* __restrict__ out, int ldo, int col0, int n_he)
{
    int wid  = (blockIdx.x * blockDim.x + threadIdx.x) >> 5;
    int lane = threadIdx.x & 31;
    if (wid >= n_he) return;
    const int h = wid;

    const float* qp = qfn + (size_t)h * 128;
    float q0 = qp[lane], q1 = qp[lane + 32], q2 = qp[lane + 64], q3 = qp[lane + 96];

    const int s = offs[h], e = offs[h + 1];

    float mx = -3.4e38f;
    for (int i = s; i < e; ++i) {
        const float* kp = k2 + (size_t)csr[i] * 128;
        float p = kp[lane] * q0 + kp[lane + 32] * q1 + kp[lane + 64] * q2 + kp[lane + 96] * q3;
        for (int o = 16; o >= 1; o >>= 1) p += __shfl_xor(p, o, 32);
        p *= 0.125f;                          // 1/sqrt(Q_DIM)
        p = (p > 0.0f) ? p : 0.01f * p;       // leaky_relu
        mx = fmaxf(mx, p);
    }

    float denom = 0.0f, a0 = 0.0f, a1 = 0.0f, a2 = 0.0f, a3 = 0.0f;
    for (int i = s; i < e; ++i) {
        int d = csr[i];
        const float* kp = k2 + (size_t)d * 128;
        float p = kp[lane] * q0 + kp[lane + 32] * q1 + kp[lane + 64] * q2 + kp[lane + 96] * q3;
        for (int o = 16; o >= 1; o >>= 1) p += __shfl_xor(p, o, 32);
        p *= 0.125f;
        p = (p > 0.0f) ? p : 0.01f * p;
        float ex = __expf(p - mx);
        denom += ex;
        const float* vp = v2 + (size_t)d * 128;
        a0 += ex * vp[lane];
        a1 += ex * vp[lane + 32];
        a2 += ex * vp[lane + 64];
        a3 += ex * vp[lane + 96];
    }
    float inv = 1.0f / fmaxf(denom, 1e-20f);  // empty segment -> 0, matches reference
    float* op = out + (size_t)h * ldo + col0;
    op[lane]      = a0 * inv;
    op[lane + 32] = a1 * inv;
    op[lane + 64] = a2 * inv;
    op[lane + 96] = a3 * inv;
}

// ---------------- host orchestration ----------------
static inline void launch_gemm(const float* A, int lda, const float* W, int ldw,
                               const float* bias, float* C, int ldc,
                               int M, int N, int K, int act, hipStream_t stream)
{
    dim3 grid((N + BN - 1) / BN, (M + BM - 1) / BM);
    gemm_wmma_bias_act<<<grid, 256, 0, stream>>>(A, lda, W, ldw, bias, C, ldc, M, N, K, act);
}

extern "C" void kernel_launch(void* const* d_in, const int* in_sizes, int n_in,
                              void* d_out, int out_size, void* d_ws, size_t ws_size,
                              hipStream_t stream)
{
    const float* he_feat = (const float*)d_in[0];
    const float* feats[3] = { (const float*)d_in[1], (const float*)d_in[2], (const float*)d_in[3] };
    const int    nt[3]    = { 167, 881, 441 };
    const int    dt[3]    = { 167, 881, 441 };
    const float* w1w = (const float*)d_in[4];  const float* w1b = (const float*)d_in[5];
    const float* w2w = (const float*)d_in[6];  const float* w2b = (const float*)d_in[7];
    // d_in[8..11] = w3/w4 : dead code (feeds only the discarded node update)
    const float* w5w[3] = { (const float*)d_in[12], (const float*)d_in[14], (const float*)d_in[16] };
    const float* w5b[3] = { (const float*)d_in[13], (const float*)d_in[15], (const float*)d_in[17] };
    const float* w6w[3] = { (const float*)d_in[18], (const float*)d_in[22], (const float*)d_in[26] };
    const float* w6b[3] = { (const float*)d_in[19], (const float*)d_in[23], (const float*)d_in[27] };
    const float* w7w[3] = { (const float*)d_in[20], (const float*)d_in[24], (const float*)d_in[28] };
    const float* w7b[3] = { (const float*)d_in[21], (const float*)d_in[25], (const float*)d_in[29] };
    const float* mlp1w = (const float*)d_in[30]; const float* mlp1b = (const float*)d_in[31];
    const float* mlp2w = (const float*)d_in[32]; const float* mlp2b = (const float*)d_in[33];
    const int* srcs[3] = { (const int*)d_in[34], (const int*)d_in[36], (const int*)d_in[38] };
    const int* dsts[3] = { (const int*)d_in[35], (const int*)d_in[37], (const int*)d_in[39] };
    const int  E = in_sizes[34];

    // ---- workspace carve ----
    char* ws = (char*)d_ws;
    auto carve = [&](size_t bytes) { void* p = (void*)ws; ws += (bytes + 255) & ~(size_t)255; return p; };
    float* w12    = (float*)carve((size_t)IN_FEAT * E_DIM * 4);        // fused w1@w2
    float* b12    = (float*)carve(E_DIM * 4);
    float* qfn    = (float*)carve((size_t)N_HE * E_DIM * 4);           // 25.6 MB (L2-resident)
    float* he_new = (float*)carve((size_t)N_HE * 3 * E_DIM * 4);       // 76.8 MB
    float* xh     = (float*)carve((size_t)N_HE * 2 * E_DIM * 4);       // 51.2 MB
    float* qt     = (float*)carve((size_t)881 * Q_DIM * 4);
    float* k2     = (float*)carve((size_t)881 * E_DIM * 4);
    float* v2     = (float*)carve((size_t)881 * E_DIM * 4);
    int*   cnt    = (int*)carve((size_t)N_HE * 4);
    int*   offs   = (int*)carve((size_t)(N_HE + 1) * 4);
    int*   cursor = (int*)carve((size_t)N_HE * 4);
    int*   csr    = (int*)carve((size_t)E * 4);

    // ---- fused q_from_nodes = he_feat @ (w1@w2) + (b1@w2 + b2) ----
    launch_gemm(w1w, HE_DIM, w2w, E_DIM, nullptr, w12, E_DIM, IN_FEAT, E_DIM, HE_DIM, 0, stream);
    fuse_bias_kernel<<<1, E_DIM, 0, stream>>>(w1b, w2w, w2b, b12);
    launch_gemm(he_feat, IN_FEAT, w12, E_DIM, b12, qfn, E_DIM, N_HE, E_DIM, IN_FEAT, 0, stream);

    const int egrid = (E + 255) / 256;
    const int zgrid = (N_HE + 255) / 256;
    const int agrid = (N_HE + 7) / 8;   // 8 waves per 256-thread block

    for (int g = 0; g < 3; ++g) {
        // node projections: q = feat@w5+b5 ; k2 = q@w6+b6 ; v2 = q@w7+b7
        launch_gemm(feats[g], dt[g], w5w[g], Q_DIM, w5b[g], qt, Q_DIM, nt[g], Q_DIM, dt[g], 0, stream);
        launch_gemm(qt, Q_DIM, w6w[g], E_DIM, w6b[g], k2, E_DIM, nt[g], E_DIM, Q_DIM, 0, stream);
        launch_gemm(qt, Q_DIM, w7w[g], E_DIM, w7b[g], v2, E_DIM, nt[g], E_DIM, Q_DIM, 0, stream);
        // CSR grouped by hyperedge (segment id = src), payload = node id (dst)
        zero_i32<<<zgrid, 256, 0, stream>>>(cnt, N_HE);
        count_edges<<<egrid, 256, 0, stream>>>(srcs[g], E, cnt);
        scan_offsets<<<1, 1024, 0, stream>>>(cnt, N_HE, offs, cursor);
        fill_csr<<<egrid, 256, 0, stream>>>(srcs[g], dsts[g], E, cursor, csr);
        // atomic-free segment softmax, one wave per hyperedge
        seg_attn_wave<<<agrid, 256, 0, stream>>>(qfn, k2, v2, offs, csr,
                                                 he_new, 3 * E_DIM, g * E_DIM, N_HE);
    }

    // ---- MLP head ----
    launch_gemm(he_new, 3 * E_DIM, mlp1w, 2 * E_DIM, mlp1b, xh, 2 * E_DIM,
                N_HE, 2 * E_DIM, 3 * E_DIM, 1, stream);
    launch_gemm(xh, 2 * E_DIM, mlp2w, E_DIM, mlp2b, (float*)d_out, E_DIM,
                N_HE, E_DIM, 2 * E_DIM, 1, stream);
}
